// MetaBind_MultiEdges_83562883711143
// MI455X (gfx1250) — matrix operand order repaired and checked by reading, not compiled, not run on hardware
//
#include <hip/hip_runtime.h>
#include <hip/hip_bf16.h>

typedef __attribute__((ext_vector_type(16))) __bf16 v16bf;
typedef __attribute__((ext_vector_type(8)))  __bf16 v8bf;
typedef __attribute__((ext_vector_type(8)))  float  v8f;
typedef __attribute__((ext_vector_type(4)))  unsigned int u32x4;
typedef __attribute__((ext_vector_type(8)))  int    i32x8;
typedef __attribute__((ext_vector_type(4)))  int    i32x4;

// ---------------------------------------------------------------------------
// A-matrix segment descriptor: value(row, col) =
//   src = (f ? f : b)[ (idx ? idx[row] : row) * width + col ]
//   if ss:   src = src*ss[col] + ss[col+128]   (BatchNorm scale/shift)
//   if relu: src = max(src, 0)
// ---------------------------------------------------------------------------
struct Seg {
    const float*  f;     // fp32 source (or null)
    const __bf16* b;     // bf16 source (or null)
    const int*    idx;   // gather indices (or null = identity)
    const float*  ss;    // scale at [c], shift at [c+128] (or null)
    int width;
    int relu;
};

// ---------------------------------------------------------------------------
// Fused gather + BN-on-load + GEMM (M x Kp) @ (Kp x 128) + bias, bf16 out,
// with per-channel sum/sumsq accumulation for the next BatchNorm.
// Block: 256 threads (8 wave32), tile = 128 rows x 128 cols.
// W pre-packed [Kp/32][128][32]: each K-tile is one contiguous 8KB block,
// DMA'd into double-buffered LDS by the Tensor Data Mover while the previous
// tile computes; A staging is software-pipelined through registers.
// ---------------------------------------------------------------------------
__global__ __launch_bounds__(256)
void k_gemm(Seg s0, Seg s1, Seg s2,
            const __bf16* __restrict__ Wp,    // packed (Kp/32)*4096 bf16
            const float*  __restrict__ bias,  // 128 or null
            __bf16*       __restrict__ out,   // M x 128
            float*        __restrict__ stat,  // sums: [c]=sum,[c+128]=sumsq (or null)
            int M, int Kp)
{
    __shared__ __bf16 lA[128 * 32];       // row-major: [row][k]  (64B rows)
    __shared__ __bf16 lW[2][128 * 32];    // packed:    [col][k]  (64B cols), 2 buffers
    __shared__ float  lred[256];

    const int t    = threadIdx.x;
    const int wave = t >> 5;
    const int lane = t & 31;
    const int half = lane >> 4;
    const int l16  = lane & 15;
    const int rowBlk = blockIdx.x * 128;

    // ---- staging-role precompute (loop invariant): gather rows per segment
    const int r     = t >> 1;
    const int cbase = (t & 1) * 16;
    const int grow  = rowBlk + r;
    const bool rowok = grow < M;
    const int b0 = s0.width, b1 = b0 + s1.width, b2 = b1 + s2.width;
    int srow0 = 0, srow1 = 0, srow2 = 0;
    if (rowok) {
        if (s0.width) srow0 = s0.idx ? s0.idx[grow] : grow;
        if (s1.width) srow1 = s1.idx ? s1.idx[grow] : grow;
        if (s2.width) srow2 = s2.idx ? s2.idx[grow] : grow;
    }

    // fetch one 16-col chunk of the A tile (gather + BN + relu) into registers
    auto fetch = [&](int k0, float* vals) {
        const int c0 = k0 + cbase;
        const Seg* sp = nullptr; int srow = 0, cl0 = 0;
        if (rowok) {
            if (c0 + 16 <= b0)                  { sp = &s0; srow = srow0; cl0 = c0; }
            else if (c0 >= b0 && c0 + 16 <= b1) { sp = &s1; srow = srow1; cl0 = c0 - b0; }
            else if (c0 >= b1 && c0 + 16 <= b2) { sp = &s2; srow = srow2; cl0 = c0 - b1; }
        }
        if (sp) {
            const long base = (long)srow * sp->width + cl0;
            if (sp->f) {
                const float* fp = sp->f + base;
#pragma unroll
                for (int i = 0; i < 16; ++i) vals[i] = fp[i];
            } else {
                const __bf16* bp = sp->b + base;
#pragma unroll
                for (int i = 0; i < 16; ++i) vals[i] = (float)bp[i];
            }
            if (sp->ss) {
                const float* ssp = sp->ss + cl0;
#pragma unroll
                for (int i = 0; i < 16; ++i) vals[i] = vals[i] * ssp[i] + ssp[i + 128];
            }
            if (sp->relu) {
#pragma unroll
                for (int i = 0; i < 16; ++i) vals[i] = fmaxf(vals[i], 0.f);
            }
        } else {
            // slow path: pad region, mixed chunk, or out-of-range row
#pragma unroll
            for (int i = 0; i < 16; ++i) {
                const int c = c0 + i;
                float v = 0.f;
                if (rowok) {
                    const Seg* q = nullptr; int qs = 0, cl = c;
                    if (c < b0)       { q = &s0; qs = srow0; }
                    else if (c < b1)  { q = &s1; qs = srow1; cl = c - b0; }
                    else if (c < b2)  { q = &s2; qs = srow2; cl = c - b1; }
                    if (q) {
                        v = q->f ? q->f[(long)qs * q->width + cl]
                                 : (float)q->b[(long)qs * q->width + cl];
                        if (q->ss)   v = v * q->ss[cl] + q->ss[cl + 128];
                        if (q->relu) v = fmaxf(v, 0.f);
                    }
                }
                vals[i] = v;
            }
        }
    };

#if __has_builtin(__builtin_amdgcn_tensor_load_to_lds)
    // issue a TDM copy of one packed 8KB W tile into an LDS buffer (wave-level)
    auto tdm = [&](int tile, __bf16* dstLds) {
        const unsigned long long ga =
            (unsigned long long)(size_t)(Wp + (size_t)tile * 4096);
        const unsigned lds = (unsigned)(size_t)(void*)dstLds;
        u32x4 g0;
        g0.x = 1u;                                        // count=1
        g0.y = lds;                                       // lds_addr
        g0.z = (unsigned)(ga & 0xFFFFFFFFull);            // global_addr lo
        g0.w = (unsigned)((ga >> 32) & 0x1FFFFFFull) | (2u << 30); // hi | type=2
        i32x8 g1;
        g1[0] = (1 << 16);                                // data_size=1 (2B)
        g1[1] = (int)(4096u << 16);                       // tensor_dim0 = 4096
        g1[2] = (int)(1u << 16);                          // tensor_dim1 = 1
        g1[3] = (int)(4096u << 16);                       // tile_dim0 = 4096
        g1[4] = 1;                                        // tile_dim1 = 1
        g1[5] = 4096;                                     // tensor_dim0_stride
        g1[6] = 0;
        g1[7] = 0;
        i32x4 gz = {0, 0, 0, 0};
#if __clang_major__ >= 23
        i32x8 gz8 = {0, 0, 0, 0, 0, 0, 0, 0};
        __builtin_amdgcn_tensor_load_to_lds(g0, g1, gz, gz, gz8, 0);
#else
        __builtin_amdgcn_tensor_load_to_lds(g0, g1, gz, gz, 0);
#endif
    };
#endif

    v8f acc[8];
#pragma unroll
    for (int i = 0; i < 8; ++i)
#pragma unroll
        for (int j = 0; j < 8; ++j) acc[i][j] = 0.f;

    const int T = Kp >> 5;               // number of 32-wide K tiles
    float vals[16];
    fetch(0, vals);                      // pipeline prologue: A chunk for tile 0
#if __has_builtin(__builtin_amdgcn_tensor_load_to_lds)
    if (wave == 0) tdm(0, lW[0]);        // W tile 0 in flight
#else
    for (int i = t; i < 32 * 128; i += 256) lW[0][i] = Wp[i];
#endif

    for (int it = 0; it < T; ++it) {
        const __bf16* lWcur = lW[it & 1];
        __bf16*       lWnxt = lW[(it & 1) ^ 1];

        // ---- publish this tile's A chunk (prev compute finished at loop-end barrier)
        {
            v8bf o0, o1;
#pragma unroll
            for (int j = 0; j < 8; ++j) { o0[j] = (__bf16)vals[j]; o1[j] = (__bf16)vals[8 + j]; }
            *(v8bf*)&lA[r * 32 + cbase]     = o0;
            *(v8bf*)&lA[r * 32 + cbase + 8] = o1;
        }
#if __has_builtin(__builtin_amdgcn_tensor_load_to_lds)
        if (wave == 0) __builtin_amdgcn_s_wait_tensorcnt(0);  // W tile `it` landed
#endif
        __syncthreads();                 // lA + lWcur visible to all waves

        // ---- start next tile's transfers (overlap with WMMA below)
        float nvals[16];
        if (it + 1 < T) {
            fetch((it + 1) * 32, nvals);
#if __has_builtin(__builtin_amdgcn_tensor_load_to_lds)
            if (wave == 0) tdm(it + 1, lWnxt);
#else
            for (int i = t; i < 32 * 128; i += 256)
                lWnxt[i] = Wp[(size_t)(it + 1) * 4096 + i];
#endif
            __builtin_prefetch(Wp + (size_t)(it + 2) * 4096, 0, 1);
        }

        // ---- fragments (contiguous 16B LDS runs) + WMMA chain ----
        // A 16x32: lane(half,l16) -> runs at k = 8*half and k = 16+8*half
        const int arow = (wave * 16 + l16) * 32;
        const v8bf a_lo = *(const v8bf*)&lA[arow + 8 * half];
        const v8bf a_hi = *(const v8bf*)&lA[arow + 16 + 8 * half];
        const v16bf afrag = __builtin_shufflevector(
            a_lo, a_hi, 0, 1, 2, 3, 4, 5, 6, 7, 8, 9, 10, 11, 12, 13, 14, 15);

        // preload all B fragments, then issue the WMMA chain (partial waits)
        v8bf blo[8], bhi[8];
#pragma unroll
        for (int nt = 0; nt < 8; ++nt) {
            const int boff = (nt * 16 + l16) * 32 + 8 * half;
            blo[nt] = *(const v8bf*)&lWcur[boff];
            bhi[nt] = *(const v8bf*)&lWcur[boff + 16];
        }
#pragma unroll
        for (int nt = 0; nt < 8; ++nt) {
            const v16bf bfrag = __builtin_shufflevector(
                blo[nt], bhi[nt], 0, 1, 2, 3, 4, 5, 6, 7, 8, 9, 10, 11, 12, 13, 14, 15);
            acc[nt] = __builtin_amdgcn_wmma_f32_16x16x32_bf16(
                false, afrag, false, bfrag, (short)0, acc[nt], false, false);
        }

        __syncthreads();                 // everyone done with lA before next store
        if (it + 1 < T) {
#pragma unroll
            for (int i = 0; i < 16; ++i) vals[i] = nvals[i];
        }
    }

    // ---- epilogue: bias, store bf16, per-channel stats ----
    lred[t] = 0.f;
    __syncthreads();

#pragma unroll
    for (int nt = 0; nt < 8; ++nt) {
        const int col = nt * 16 + l16;
        const float bv = bias ? bias[col] : 0.f;
#pragma unroll
        for (int rr = 0; rr < 8; ++rr) {
            const int orow = rowBlk + wave * 16 + half * 8 + rr;  // D: M=r+8*half
            const float v = acc[nt][rr] + bv;
            if (orow < M) {
                out[(long)orow * 128 + col] = (__bf16)v;
                if (stat) {
                    atomicAdd(&lred[col], v);
                    atomicAdd(&lred[col + 128], v * v);
                }
            }
        }
    }
    __syncthreads();
    if (stat) atomicAdd(&stat[t], lred[t]);
}

// ---------------------------------------------------------------------------
// Support kernels
// ---------------------------------------------------------------------------
__global__ void k_zero(float* p, long n) {
    long i = blockIdx.x * (long)blockDim.x + threadIdx.x;
    if (i < n) p[i] = 0.f;
}

// pack fp32 (K x 128) -> bf16 [Kp/32][128][32], zero-padded K..Kp
__global__ void k_pack(const float* w, __bf16* o, int K, int Kp) {
    int i = blockIdx.x * blockDim.x + threadIdx.x;    // over Kp*128
    if (i < Kp * 128) {
        int k = i >> 7, n = i & 127;
        float v = (k < K) ? w[i] : 0.f;
        o[(size_t)(k >> 5) * 4096 + n * 32 + (k & 31)] = (__bf16)v;
    }
}

__global__ void k_edge_attr(const float* __restrict__ pos,
                            const int* __restrict__ src, const int* __restrict__ dst,
                            float* __restrict__ ea, long E) {
    long e = blockIdx.x * (long)blockDim.x + threadIdx.x;
    if (e >= E) return;
    int s = src[e], d = dst[e];
    float sx = pos[s*3], sy = pos[s*3+1], sz = pos[s*3+2];
    float dx = pos[d*3], dy = pos[d*3+1], dz = pos[d*3+2];
    float ux = sx-dx+1e-6f, uy = sy-dy+1e-6f, uz = sz-dz+1e-6f;
    float dist = sqrtf(ux*ux + uy*uy + uz*uz) * 0.1f;                 // /R
    float dot  = sx*dx + sy*dy + sz*dz;
    float nn   = sqrtf(sx*sx+sy*sy+sz*sz) * sqrtf(dx*dx+dy*dy+dz*dz);
    float cosv = dot / fmaxf(nn, 1e-8f);
    ea[e*2+0] = dist;
    ea[e*2+1] = (cosv + 1.f) * 0.5f;
}

// column sum / sumsq over M rows, C<=128 cols (fp32 input)
__global__ void k_colstats(const float* __restrict__ a, long M, int C,
                           float* __restrict__ sums) {
    __shared__ float s[256];
    int t = threadIdx.x;
    if (t < 2 * C) s[t] = 0.f;
    __syncthreads();
    long r = blockIdx.x * (long)blockDim.x + t;
    if (r < M) {
        for (int c = 0; c < C; ++c) {
            float v = a[r * C + c];
            atomicAdd(&s[c], v);
            atomicAdd(&s[c + C], v * v);
        }
    }
    __syncthreads();
    if (t < C) {
        atomicAdd(&sums[t], s[t]);
        atomicAdd(&sums[t + 128], s[t + C]);
    }
}

// sums -> (scale, shift) with gamma/beta; var = E[x^2]-E[x]^2 (biased, as jnp.var)
__global__ void k_fin(const float* __restrict__ sums,
                      const float* __restrict__ g, const float* __restrict__ b,
                      float minv, int C, float* __restrict__ ss) {
    int c = threadIdx.x;
    if (c < C) {
        float mu  = sums[c] * minv;
        float var = sums[c + 128] * minv - mu * mu;
        float sc  = rsqrtf(var + 1e-5f);
        if (g) sc *= g[c];
        float sh = (b ? b[c] : 0.f) - mu * sc;
        ss[c] = sc;
        ss[c + 128] = sh;
    }
}

// acc[idx[r]*128+c] += bn(h[r*128+c]);  grid = M blocks x 128 threads
__global__ void k_segsum(const __bf16* __restrict__ h, const float* __restrict__ ss,
                         const int* __restrict__ idx, float* __restrict__ acc, long M) {
    long r = blockIdx.x;
    int  c = threadIdx.x;
    if (r < M) {
        float v = (float)h[r * 128 + c] * ss[c] + ss[c + 128];
        atomicAdd(&acc[(long)idx[r] * 128 + c], v);
    }
}

// write post-BN values into a strided slice (u_cat builder)
__global__ void k_bnw(const __bf16* __restrict__ h, const float* __restrict__ ss,
                      __bf16* __restrict__ dstp, int M, int ld) {
    int r = blockIdx.x, c = threadIdx.x;
    if (r < M)
        dstp[r * ld + c] = (__bf16)((float)h[r * 128 + c] * ss[c] + ss[c + 128]);
}

// out[g] = bs2 + sum_c relu(bn(h1[g,c])) * Ws2[c]
__global__ void k_score(const __bf16* __restrict__ h, const float* __restrict__ ss,
                        const float* __restrict__ W2, const float* __restrict__ b2,
                        float* __restrict__ out, int G) {
    int g = blockIdx.x * blockDim.x + threadIdx.x;
    if (g < G) {
        float a = b2[0];
        for (int c = 0; c < 128; ++c) {
            float v = (float)h[g * 128 + c] * ss[c] + ss[c + 128];
            a += fmaxf(v, 0.f) * W2[c];
        }
        out[g] = a;
    }
}

// ---------------------------------------------------------------------------
// Host orchestration
// ---------------------------------------------------------------------------
extern "C" void kernel_launch(void* const* d_in, const int* in_sizes, int n_in,
                              void* d_out, int out_size, void* d_ws, size_t ws_size,
                              hipStream_t stream) {
    (void)n_in; (void)ws_size;
    auto P = [&](int i) { return (const float*)d_in[i]; };

    const float* x     = P(0);
    const float* pos   = P(1);
    const int*   batch = (const int*)d_in[2];
    const int*   eidx  = (const int*)d_in[3];
    const long N = in_sizes[0] / 64;
    const long E = in_sizes[3] / 2;
    const int  G = out_size;
    const int* src = eidx;
    const int* dst = eidx + E;

    // ---- workspace bump allocator ----
    char* wsp = (char*)d_ws;
    size_t off = 0;
    auto alloc = [&](size_t bytes) -> void* {
        void* p = wsp + off;
        off += (bytes + 255) & ~(size_t)255;
        return p;
    };
    float*  ea        = (float*) alloc((size_t)E * 2 * 4);
    __bf16* eA        = (__bf16*)alloc((size_t)E * 128 * 2);
    __bf16* eS        = (__bf16*)alloc((size_t)E * 128 * 2);
    __bf16* xh        = (__bf16*)alloc((size_t)N * 128 * 2);
    __bf16* xS        = (__bf16*)alloc((size_t)N * 128 * 2);
    float*  node_acc  = (float*) alloc((size_t)N * 128 * 4);
    float*  graph_acc = (float*) alloc((size_t)G * 128 * 4);
    __bf16* uA        = (__bf16*)alloc((size_t)G * 128 * 2);
    __bf16* uS        = (__bf16*)alloc((size_t)G * 128 * 2);
    __bf16* ucat      = (__bf16*)alloc((size_t)G * 384 * 2);
    __bf16* sh1       = (__bf16*)alloc((size_t)G * 128 * 2);
    float*  stats     = (float*) alloc((size_t)24 * 512 * 4);
    auto SUMS = [&](int i) { return stats + i * 512; };
    auto SS   = [&](int i) { return stats + i * 512 + 256; };

    auto zero = [&](float* p, long n) {
        k_zero<<<(unsigned)((n + 255) / 256), 256, 0, stream>>>(p, n);
    };
    auto packW = [&](const float* w, int K, int Kp) -> __bf16* {
        __bf16* o = (__bf16*)alloc((size_t)Kp * 128 * 2);
        k_pack<<<(Kp * 128 + 255) / 256, 256, 0, stream>>>(w, o, K, Kp);
        return o;
    };
    auto gemm = [&](Seg a, Seg b, Seg c, const __bf16* W, const float* bias,
                    __bf16* o, float* sums, long M, int Kp) {
        k_gemm<<<(unsigned)((M + 127) / 128), 256, 0, stream>>>(
            a, b, c, W, bias, o, sums, (int)M, Kp);
    };
    auto fin = [&](int slot, const float* g, const float* b, float minv, int C) {
        k_fin<<<1, 128, 0, stream>>>(SUMS(slot), g, b, minv, C, SS(slot));
    };
    auto SegF = [&](const float* f, const int* idx, const float* ss, int w, int relu) {
        Seg s{f, nullptr, idx, ss, w, relu}; return s;
    };
    auto SegB = [&](const __bf16* bb, const int* idx, const float* ss, int w, int relu) {
        Seg s{nullptr, bb, idx, ss, w, relu}; return s;
    };
    Seg Z{nullptr, nullptr, nullptr, nullptr, 0, 0};

    // ---- pack weights (pad K to multiple of 32) ----
    __bf16* W_ee1 = packW(P(8),  130, 160);
    __bf16* W_ee2 = packW(P(12), 128, 128);
    __bf16* W_en1 = packW(P(16), 192, 192);
    __bf16* W_en2 = packW(P(20), 128, 128);
    __bf16* W_eg1 = packW(P(24), 128, 128);
    __bf16* W_eg2 = packW(P(28), 128, 128);
    __bf16* W_se1 = packW(P(32), 384, 384);
    __bf16* W_se2 = packW(P(36), 128, 128);
    __bf16* W_sn1 = packW(P(40), 256, 256);
    __bf16* W_sn2 = packW(P(44), 128, 128);
    __bf16* W_sg1 = packW(P(48), 256, 256);
    __bf16* W_sg2 = packW(P(52), 128, 128);
    __bf16* W_sc1 = packW(P(56), 384, 384);

    zero(stats, 24 * 512);

    // ---- input BNs ----
    k_colstats<<<(unsigned)((N + 255) / 256), 256, 0, stream>>>(x, N, 64, SUMS(0));
    fin(0, P(4), P(5), 1.f / (float)N, 64);
    k_edge_attr<<<(unsigned)((E + 255) / 256), 256, 0, stream>>>(pos, src, dst, ea, E);
    k_colstats<<<(unsigned)((E + 255) / 256), 256, 0, stream>>>(ea, E, 2, SUMS(1));
    fin(1, P(6), P(7), 1.f / (float)E, 2);

    // ---- encoder: edge ----
    gemm(SegF(x, src, SS(0), 64, 0), SegF(x, dst, SS(0), 64, 0),
         SegF(ea, nullptr, SS(1), 2, 0), W_ee1, P(9), eS, SUMS(2), E, 160);
    fin(2, P(10), P(11), 1.f / (float)E, 128);
    gemm(SegB(eS, nullptr, SS(2), 128, 1), Z, Z, W_ee2, P(13), eA, SUMS(3), E, 128);
    fin(3, P(14), P(15), 1.f / (float)E, 128);

    // ---- encoder: node ----
    zero(node_acc, N * 128);
    k_segsum<<<(unsigned)E, 128, 0, stream>>>(eA, SS(3), dst, node_acc, E);
    gemm(SegF(x, nullptr, SS(0), 64, 0), SegF(node_acc, nullptr, nullptr, 128, 0), Z,
         W_en1, P(17), xS, SUMS(4), N, 192);
    fin(4, P(18), P(19), 1.f / (float)N, 128);
    gemm(SegB(xS, nullptr, SS(4), 128, 1), Z, Z, W_en2, P(21), xh, SUMS(5), N, 128);
    fin(5, P(22), P(23), 1.f / (float)N, 128);

    // ---- encoder: global ----
    zero(graph_acc, (long)G * 128);
    k_segsum<<<(unsigned)N, 128, 0, stream>>>(xh, SS(5), batch, graph_acc, N);
    gemm(SegF(graph_acc, nullptr, nullptr, 128, 0), Z, Z, W_eg1, P(25), uS, SUMS(6), G, 128);
    fin(6, P(26), P(27), 1.f / (float)G, 128);
    gemm(SegB(uS, nullptr, SS(6), 128, 1), Z, Z, W_eg2, P(29), uA, SUMS(7), G, 128);
    fin(7, P(30), P(31), 1.f / (float)G, 128);
    k_bnw<<<G, 128, 0, stream>>>(uA, SS(7), ucat + 0, G, 384);

    // ---- 2 GN steps ----
    int cur_e = 3, cur_x = 5, cur_u = 7;
    for (int s = 0; s < 2; ++s) {
        const int b = 8 + s * 6;
        gemm(SegB(xh, src, SS(cur_x), 128, 0), SegB(xh, dst, SS(cur_x), 128, 0),
             SegB(eA, nullptr, SS(cur_e), 128, 0), W_se1, P(33), eS, SUMS(b), E, 384);
        fin(b, P(34), P(35), 1.f / (float)E, 128);
        gemm(SegB(eS, nullptr, SS(b), 128, 1), Z, Z, W_se2, P(37), eA, SUMS(b + 1), E, 128);
        fin(b + 1, P(38), P(39), 1.f / (float)E, 128);
        cur_e = b + 1;

        zero(node_acc, N * 128);
        k_segsum<<<(unsigned)E, 128, 0, stream>>>(eA, SS(cur_e), dst, node_acc, E);
        gemm(SegB(xh, nullptr, SS(cur_x), 128, 0),
             SegF(node_acc, nullptr, nullptr, 128, 0), Z,
             W_sn1, P(41), xS, SUMS(b + 2), N, 256);
        fin(b + 2, P(42), P(43), 1.f / (float)N, 128);
        gemm(SegB(xS, nullptr, SS(b + 2), 128, 1), Z, Z, W_sn2, P(45), xh, SUMS(b + 3), N, 128);
        fin(b + 3, P(46), P(47), 1.f / (float)N, 128);
        cur_x = b + 3;

        zero(graph_acc, (long)G * 128);
        k_segsum<<<(unsigned)N, 128, 0, stream>>>(xh, SS(cur_x), batch, graph_acc, N);
        gemm(SegB(uA, nullptr, SS(cur_u), 128, 0),
             SegF(graph_acc, nullptr, nullptr, 128, 0), Z,
             W_sg1, P(49), uS, SUMS(b + 4), G, 256);
        fin(b + 4, P(50), P(51), 1.f / (float)G, 128);
        gemm(SegB(uS, nullptr, SS(b + 4), 128, 1), Z, Z, W_sg2, P(53), uA, SUMS(b + 5), G, 128);
        fin(b + 5, P(54), P(55), 1.f / (float)G, 128);
        cur_u = b + 5;
        k_bnw<<<G, 128, 0, stream>>>(uA, SS(cur_u), ucat + (s + 1) * 128, G, 384);
    }

    // ---- score head ----
    gemm(SegB(ucat, nullptr, nullptr, 384, 0), Z, Z, W_sc1, nullptr, sh1, SUMS(20), G, 384);
    fin(20, P(57), P(58), 1.f / (float)G, 128);
    k_score<<<1, 64, 0, stream>>>(sh1, SS(20), P(59), P(60), (float*)d_out, G);
}
